// WishfulThinking_JK_33200097198308
// MI455X (gfx1250) — compile-verified, matrix operand order checked by reading, not run on hardware
//
#include <hip/hip_runtime.h>
#include <hip/hip_bf16.h>
#include <stdint.h>
#include <stddef.h>

// ---------------------------------------------------------------------------
// GATv2 x3 + add-pool + MLP head for MI455X (gfx1250, wave32, WMMA).
// Dense node GEMMs run as bf16 WMMA (v_wmma_f32_16x16x32_bf16, f32 accum);
// edge softmax / aggregation phases are bandwidth-bound f32 + atomics.
// ---------------------------------------------------------------------------

typedef __attribute__((ext_vector_type(16))) __bf16 v16bf;
typedef __attribute__((ext_vector_type(8)))  __bf16 v8bf;
typedef __attribute__((ext_vector_type(8)))  float  v8f;

#define NN        20000
#define EFULL     160000
#define ELIG      80000
#define NGRAPH    64
#define DEDGE     16

// ---- helpers --------------------------------------------------------------

static __device__ __forceinline__ unsigned short f32_to_bf16_rne(float f) {
  unsigned u = __float_as_uint(f);
  unsigned r = u + 0x7FFFu + ((u >> 16) & 1u);
  return (unsigned short)(r >> 16);
}

// order-preserving float <-> uint key for atomicMax-based segment max
static __device__ __forceinline__ unsigned fkey(float f) {
  unsigned u = __float_as_uint(f);
  return (u & 0x80000000u) ? ~u : (u | 0x80000000u);
}
static __device__ __forceinline__ float funkey(unsigned k) {
  unsigned u = (k & 0x80000000u) ? (k & 0x7FFFFFFFu) : ~k;
  return __uint_as_float(u);
}

// ---- generic fill ---------------------------------------------------------

__global__ void k_fill_u32(unsigned* __restrict__ p, unsigned v, int n) {
  int i = blockIdx.x * blockDim.x + threadIdx.x;
  if (i < n) p[i] = v;
}

// ---- bf16 conversion / weight transpose -----------------------------------

__global__ void k_f32_to_bf16(const float* __restrict__ x,
                              unsigned short* __restrict__ y, int n) {
  int i = blockIdx.x * blockDim.x + threadIdx.x;
  if (i < n) y[i] = f32_to_bf16_rne(x[i]);
}

// W: [K,N] f32 row-major  ->  Wt: [N,K] bf16 (so B-fragment loads are contiguous)
__global__ void k_transpose_bf16(const float* __restrict__ W,
                                 unsigned short* __restrict__ Wt, int K, int N) {
  int i = blockIdx.x * blockDim.x + threadIdx.x;
  if (i >= N * K) return;
  int n = i / K, k = i - n * K;
  Wt[i] = f32_to_bf16_rne(W[(size_t)k * N + n]);
}

// ---- bf16 WMMA GEMM: C[M,N] = A[M,K] * Bt[N,K]^T + bias ------------------
// One wave per 16x16 C tile; 4 waves/block cover a 16x64 strip.
// A fragment (16x32 bf16): lane L<16 = row M=L holds K in {0..7,16..23};
// lane L+16 = row M=L holds K in {8..15,24..31}. B mirrors A over Bt rows.
// C/D: lane L<16 -> N=L, vgpr j -> M=j ; lane L+16 -> N=L-16, vgpr j -> M=8+j.

__global__ void __launch_bounds__(128)
k_gemm_bf16_wmma(const unsigned short* __restrict__ A,   // [M,K] bf16
                 const unsigned short* __restrict__ Bt,  // [N,K] bf16
                 const float* __restrict__ bias,         // [N]
                 float* __restrict__ C,                  // [M,N]
                 int M, int K, int N) {
  int wave = threadIdx.x >> 5;
  int lane = threadIdx.x & 31;
  int n0 = blockIdx.x * 64 + wave * 16;
  int m0 = blockIdx.y * 16;
  if (n0 >= N || m0 >= M) return;
  int r  = lane & 15;
  int hi = lane >> 4;

  const __bf16* a_row = reinterpret_cast<const __bf16*>(A)  + (size_t)(m0 + r) * K + hi * 8;
  const __bf16* b_row = reinterpret_cast<const __bf16*>(Bt) + (size_t)(n0 + r) * K + hi * 8;

  v8f acc = {};
  for (int k0 = 0; k0 < K; k0 += 32) {
    v8bf alo = *reinterpret_cast<const v8bf*>(a_row + k0);
    v8bf ahi = *reinterpret_cast<const v8bf*>(a_row + k0 + 16);
    v8bf blo = *reinterpret_cast<const v8bf*>(b_row + k0);
    v8bf bhi = *reinterpret_cast<const v8bf*>(b_row + k0 + 16);
    v16bf a, b;
#pragma unroll
    for (int i = 0; i < 8; ++i) {
      a[i] = alo[i]; a[8 + i] = ahi[i];
      b[i] = blo[i]; b[8 + i] = bhi[i];
    }
    acc = __builtin_amdgcn_wmma_f32_16x16x32_bf16(
        /*neg_a=*/false, a, /*neg_b=*/false, b,
        /*c_mod=*/(short)0, acc, /*reuse_a=*/false, /*reuse_b=*/false);
  }

  float bv = bias ? bias[n0 + r] : 0.0f;
#pragma unroll
  for (int j = 0; j < 8; ++j)
    C[(size_t)(m0 + hi * 8 + j) * N + (n0 + r)] = acc[j] + bv;
}

// ---- self-loop edge_attr mean (PyG fill_value='mean') ----------------------

__global__ void k_ea_scatter(const int* __restrict__ ei, const float* __restrict__ ea,
                             float* __restrict__ easum, float* __restrict__ deg, int E) {
  int e = blockIdx.x * blockDim.x + threadIdx.x;
  if (e >= E) return;
  int dst = ei[E + e];
#pragma unroll
  for (int k = 0; k < DEDGE; ++k)
    atomicAdd(&easum[(size_t)dst * DEDGE + k], ea[(size_t)e * DEDGE + k]);
  atomicAdd(&deg[dst], 1.0f);
}

__global__ void k_ea_div(const float* __restrict__ easum, const float* __restrict__ deg,
                         float* __restrict__ ealoop, int n) {
  int i = blockIdx.x * blockDim.x + threadIdx.x;
  if (i >= n) return;
  int node = i / DEDGE;
  ealoop[i] = easum[i] / fmaxf(deg[node], 1.0f);
}

// ---- attention score: alpha[e,h] = att_h . lrelu(xl[src]+xr[dst]+ea@We) ----
// one wave per (edge,head); edges e>=E are self loops (src=dst=e-E, ea=ealoop)

__global__ void __launch_bounds__(256)
k_edge_alpha(const int* __restrict__ ei, const float* __restrict__ ea,
             const float* __restrict__ ealoop,
             const float* __restrict__ xl, const float* __restrict__ xr,
             const float* __restrict__ We, const float* __restrict__ att,
             float* __restrict__ alpha, unsigned* __restrict__ amax,
             int E, int H, int C) {
  int Dout = H * C;
  int w = blockIdx.x * (blockDim.x >> 5) + (threadIdx.x >> 5);
  int lane = threadIdx.x & 31;
  int total = (E + NN) * H;
  if (w >= total) return;
  int e = w / H, h = w - e * H;
  int src, dst; const float* eap;
  if (e < E) { src = ei[e]; dst = ei[E + e]; eap = ea + (size_t)e * DEDGE; }
  else       { src = dst = e - E;            eap = ealoop + (size_t)src * DEDGE; }

  float ek[DEDGE];
#pragma unroll
  for (int k = 0; k < DEDGE; ++k) ek[k] = eap[k];

  const float* xls  = xl + (size_t)src * Dout + h * C;
  const float* xrd  = xr + (size_t)dst * Dout + h * C;
  const float* atth = att + h * C;

  float part = 0.f;
  for (int c = lane; c < C; c += 32) {
    float eec = 0.f;
#pragma unroll
    for (int k = 0; k < DEDGE; ++k) eec += ek[k] * We[(size_t)k * Dout + h * C + c];
    float m = xls[c] + xrd[c] + eec;
    m = (m > 0.f) ? m : 0.2f * m;          // leaky_relu 0.2
    part += atth[c] * m;
  }
#pragma unroll
  for (int off = 16; off; off >>= 1) part += __shfl_xor(part, off, 32);
  if (lane == 0) {
    alpha[w] = part;
    atomicMax(&amax[(size_t)dst * H + h], fkey(part));
  }
}

// ---- segment softmax: ex = exp(alpha - max), denom = segment_sum(ex) -------

__global__ void k_edge_exp(const int* __restrict__ ei, const float* __restrict__ alpha,
                           const unsigned* __restrict__ amax,
                           float* __restrict__ ex, float* __restrict__ denom,
                           int E, int H) {
  int i = blockIdx.x * blockDim.x + threadIdx.x;
  int total = (E + NN) * H;
  if (i >= total) return;
  int e = i / H, h = i - e * H;
  int dst = (e < E) ? ei[E + e] : (e - E);
  float v = __expf(alpha[i] - funkey(amax[(size_t)dst * H + h]));
  ex[i] = v;
  atomicAdd(&denom[(size_t)dst * H + h], v);
}

// ---- weighted aggregation: out[dst] += a * xl[src] -------------------------

__global__ void __launch_bounds__(256)
k_edge_agg(const int* __restrict__ ei, const float* __restrict__ ex,
           const float* __restrict__ denom, const float* __restrict__ xl,
           float* __restrict__ out, int E, int H, int C) {
  int Dout = H * C;
  int w = blockIdx.x * (blockDim.x >> 5) + (threadIdx.x >> 5);
  int lane = threadIdx.x & 31;
  int total = (E + NN) * H;
  if (w >= total) return;
  int e = w / H, h = w - e * H;
  int src, dst;
  if (e < E) { src = ei[e]; dst = ei[E + e]; }
  else       { src = dst = e - E; }
  float a = ex[w] / (denom[(size_t)dst * H + h] + 1e-16f);
  const float* xls = xl + (size_t)src * Dout + h * C;
  float* od = out + (size_t)dst * Dout + h * C;
  for (int c = lane; c < C; c += 32)
    atomicAdd(&od[c], xls[c] * a);
}

__global__ void k_bias_relu(float* __restrict__ out, const float* __restrict__ bias,
                            int n, int Dout) {
  int i = blockIdx.x * blockDim.x + threadIdx.x;
  if (i >= n) return;
  float v = out[i] + bias[i % Dout];
  out[i] = v > 0.f ? v : 0.f;
}

// ---- global add pool -------------------------------------------------------

__global__ void k_pool(const float* __restrict__ h, const int* __restrict__ batch,
                       float* __restrict__ pool, int D, int stride, int off) {
  int i = blockIdx.x * blockDim.x + threadIdx.x;
  if (i >= NN * D) return;
  int node = i / D, c = i - node * D;
  atomicAdd(&pool[(size_t)batch[node] * stride + off + c], h[i]);
}

// ---- MLP head --------------------------------------------------------------

__global__ void __launch_bounds__(64)
k_fc1(const float* __restrict__ z, const float* __restrict__ w,
      const float* __restrict__ b, float* __restrict__ o) {
  int row = blockIdx.x;      // 0..63 graphs
  int col = threadIdx.x;     // 0..63
  float s = b[col];
  for (int k = 0; k < 768; ++k) s += z[(size_t)row * 768 + k] * w[(size_t)k * 64 + col];
  o[(size_t)row * 64 + col] = s > 0.f ? s : 0.f;
}

__global__ void __launch_bounds__(64)
k_fc2(const float* __restrict__ hdn, const float* __restrict__ w,
      const float* __restrict__ b, float* __restrict__ out) {
  int row = threadIdx.x;     // 0..63
  float s = b[0];
  for (int k = 0; k < 64; ++k) s += hdn[(size_t)row * 64 + k] * w[k];
  out[row] = s;
}

// ---- host-side layer driver ------------------------------------------------

namespace {
struct Scratch {
  unsigned short *xbf, *wlt, *wrt;
  float *xl, *xr, *easum, *deg, *ealoop, *alpha, *exb, *denom;
  unsigned *amax;
};

static void run_gatv2(hipStream_t s, const float* x, int Din, int H, int C,
                      const int* ei, const float* ea, int E,
                      const float* Wl, const float* bl,
                      const float* Wr, const float* br,
                      const float* We, const float* att, const float* bias,
                      float* out, const Scratch& sc) {
  const int Dout = H * C;
  const int TB = 256;
  int nx = NN * Din;
  k_f32_to_bf16<<<(nx + TB - 1) / TB, TB, 0, s>>>(x, sc.xbf, nx);
  int nw = Dout * Din;
  k_transpose_bf16<<<(nw + TB - 1) / TB, TB, 0, s>>>(Wl, sc.wlt, Din, Dout);
  k_transpose_bf16<<<(nw + TB - 1) / TB, TB, 0, s>>>(Wr, sc.wrt, Din, Dout);

  dim3 gg(Dout / 64, NN / 16);   // NN = 1250*16, Dout in {256,512}
  k_gemm_bf16_wmma<<<gg, 128, 0, s>>>(sc.xbf, sc.wlt, bl, sc.xl, NN, Din, Dout);
  k_gemm_bf16_wmma<<<gg, 128, 0, s>>>(sc.xbf, sc.wrt, br, sc.xr, NN, Din, Dout);

  k_fill_u32<<<(NN * DEDGE + TB - 1) / TB, TB, 0, s>>>((unsigned*)sc.easum, 0u, NN * DEDGE);
  k_fill_u32<<<(NN + TB - 1) / TB, TB, 0, s>>>((unsigned*)sc.deg, 0u, NN);
  k_ea_scatter<<<(E + TB - 1) / TB, TB, 0, s>>>(ei, ea, sc.easum, sc.deg, E);
  k_ea_div<<<(NN * DEDGE + TB - 1) / TB, TB, 0, s>>>(sc.easum, sc.deg, sc.ealoop, NN * DEDGE);

  int total = (E + NN) * H;
  k_fill_u32<<<(NN * H + TB - 1) / TB, TB, 0, s>>>(sc.amax, 0u, NN * H);   // key 0 == -inf
  k_fill_u32<<<(NN * H + TB - 1) / TB, TB, 0, s>>>((unsigned*)sc.denom, 0u, NN * H);

  int wpb = TB / 32;                             // 8 waves / block
  int nb = (total + wpb - 1) / wpb;
  k_edge_alpha<<<nb, TB, 0, s>>>(ei, ea, sc.ealoop, sc.xl, sc.xr, We, att,
                                 sc.alpha, sc.amax, E, H, C);
  k_edge_exp<<<(total + TB - 1) / TB, TB, 0, s>>>(ei, sc.alpha, sc.amax,
                                                  sc.exb, sc.denom, E, H);
  k_fill_u32<<<(NN * Dout + TB - 1) / TB, TB, 0, s>>>((unsigned*)out, 0u, NN * Dout);
  k_edge_agg<<<nb, TB, 0, s>>>(ei, sc.exb, sc.denom, sc.xl, out, E, H, C);
  k_bias_relu<<<(NN * Dout + TB - 1) / TB, TB, 0, s>>>(out, bias, NN * Dout, Dout);
}
} // namespace

// ---- entry -----------------------------------------------------------------

extern "C" void kernel_launch(void* const* d_in, const int* in_sizes, int n_in,
                              void* d_out, int out_size, void* d_ws, size_t ws_size,
                              hipStream_t stream) {
  (void)in_sizes; (void)n_in; (void)out_size; (void)ws_size;

  const float* x     = (const float*)d_in[0];
  const int*   ei_l  = (const int*)d_in[1];
  const float* ea_l  = (const float*)d_in[2];
  const int*   ei_f  = (const int*)d_in[3];
  const float* ea_f  = (const float*)d_in[4];
  const int*   batch = (const int*)d_in[5];
  const float *Wl1=(const float*)d_in[6],  *bl1=(const float*)d_in[7],
              *Wr1=(const float*)d_in[8],  *br1=(const float*)d_in[9],
              *We1=(const float*)d_in[10], *att1=(const float*)d_in[11],
              *bias1=(const float*)d_in[12];
  const float *Wl2=(const float*)d_in[13], *bl2=(const float*)d_in[14],
              *Wr2=(const float*)d_in[15], *br2=(const float*)d_in[16],
              *We2=(const float*)d_in[17], *att2=(const float*)d_in[18],
              *bias2=(const float*)d_in[19];
  const float *Wl3=(const float*)d_in[20], *bl3=(const float*)d_in[21],
              *Wr3=(const float*)d_in[22], *br3=(const float*)d_in[23],
              *We3=(const float*)d_in[24], *att3=(const float*)d_in[25],
              *bias3=(const float*)d_in[26];
  const float *fc1_w=(const float*)d_in[27], *fc1_b=(const float*)d_in[28],
              *fc2_w=(const float*)d_in[29], *fc2_b=(const float*)d_in[30];
  float* out = (float*)d_out;

  // workspace carve-out (256B aligned slices)
  char* ws = (char*)d_ws;
  size_t off = 0;
  auto carve = [&](size_t bytes) -> char* {
    char* p = ws + off;
    off += (bytes + 255) & ~(size_t)255;
    return p;
  };
  unsigned short* XBF = (unsigned short*)carve((size_t)NN * 512 * 2);
  unsigned short* WLT = (unsigned short*)carve((size_t)512 * 512 * 2);
  unsigned short* WRT = (unsigned short*)carve((size_t)512 * 512 * 2);
  float* XL    = (float*)carve((size_t)NN * 512 * 4);
  float* XR    = (float*)carve((size_t)NN * 512 * 4);
  float* H1    = (float*)carve((size_t)NN * 256 * 4);
  float* H2    = (float*)carve((size_t)NN * 512 * 4);
  float* H3    = (float*)carve((size_t)NN * 256 * 4);
  float* EASUM = (float*)carve((size_t)NN * DEDGE * 4);
  float* DEG   = (float*)carve((size_t)NN * 4);
  float* EALP  = (float*)carve((size_t)NN * DEDGE * 4);
  float* ALPHA = (float*)carve((size_t)(EFULL + NN) * 4 * 4);
  float* EXB   = (float*)carve((size_t)(EFULL + NN) * 4 * 4);
  unsigned* AMAX = (unsigned*)carve((size_t)NN * 4 * 4);
  float* DENOM = (float*)carve((size_t)NN * 4 * 4);
  float* POOL  = (float*)carve((size_t)NGRAPH * 768 * 4);
  float* FC1O  = (float*)carve((size_t)NGRAPH * 64 * 4);

  Scratch sc{XBF, WLT, WRT, XL, XR, EASUM, DEG, EALP, ALPHA, EXB, DENOM, AMAX};

  // conv1: in 256 -> heads 1 x 256
  run_gatv2(stream, x,  256, 1, 256, ei_l, ea_l, ELIG,
            Wl1, bl1, Wr1, br1, We1, att1, bias1, H1, sc);
  // conv2: in 256 -> heads 4 x 128
  run_gatv2(stream, H1, 256, 4, 128, ei_f, ea_f, EFULL,
            Wl2, bl2, Wr2, br2, We2, att2, bias2, H2, sc);
  // conv3: in 512 -> heads 4 x 64
  run_gatv2(stream, H2, 512, 4, 64, ei_f, ea_f, EFULL,
            Wl3, bl3, Wr3, br3, We3, att3, bias3, H3, sc);

  const int TB = 256;
  k_fill_u32<<<(NGRAPH * 768 + TB - 1) / TB, TB, 0, stream>>>((unsigned*)POOL, 0u, NGRAPH * 768);
  k_pool<<<((size_t)NN * 512 + TB - 1) / TB, TB, 0, stream>>>(H2, batch, POOL, 512, 768, 0);
  k_pool<<<((size_t)NN * 256 + TB - 1) / TB, TB, 0, stream>>>(H3, batch, POOL, 256, 768, 512);

  k_fc1<<<NGRAPH, 64, 0, stream>>>(POOL, fc1_w, fc1_b, FC1O);
  k_fc2<<<1, 64, 0, stream>>>(FC1O, fc2_w, fc2_b, out);
}
